// GAT_Encoder_79834852098591
// MI455X (gfx1250) — matrix operand order repaired and checked by reading, not compiled, not run on hardware
//
#include <hip/hip_runtime.h>
#include <hip/hip_bf16.h>

// ---------------- problem constants ----------------
#define N_NODES   16384            // B*SEQ
#define DIM       128
#define SEQ_LEN   128
#define EDGES     262144
#define EL        (EDGES + N_NODES)   // edges + self loops = 278528
#define HC1       256              // heads*hid layer1
#define HIDC      64               // hid
#define SQRT_D    11.313708498984761f

typedef __attribute__((ext_vector_type(16))) __bf16 v16bf;
typedef __attribute__((ext_vector_type(8)))  float  v8f;

// ---------------- helpers ----------------
__device__ __forceinline__ void atomicMaxF(float* addr, float v) {
    // sign-trick float max: storage initialized to -INF (0xFF800000)
    if (v >= 0.0f) atomicMax((int*)addr, __float_as_int(v));
    else           atomicMin((unsigned int*)addr, __float_as_uint(v));
}

__device__ __forceinline__ void edge_sd(const int* __restrict__ ei, int e, int& s, int& d) {
    if (e < EDGES) { s = ei[e]; d = ei[EDGES + e]; }
    else           { s = e - EDGES; d = s; }          // appended self-loops
}

// ---------------- prep kernels ----------------
__global__ __launch_bounds__(256)
void fill_f32(float* __restrict__ p, float v, int n) {
    int i = blockIdx.x * 256 + threadIdx.x;
    if (i < n) p[i] = v;
}

// h0 = bf16( x*sqrt(D) + pe[pos] )
__global__ __launch_bounds__(256)
void prep_h0(const float* __restrict__ x, const float* __restrict__ pe,
             __bf16* __restrict__ h0) {
    int i = blockIdx.x * 256 + threadIdx.x;          // i < N*D
    int d = i & (DIM - 1);
    int pos = (i >> 7) & (SEQ_LEN - 1);
    float v = x[i] * SQRT_D + pe[(pos << 7) + d];
    h0[i] = (__bf16)v;
}

// Wt[n][k] = bf16(W[k][n])   (rows = K, cols = Ncols)
__global__ __launch_bounds__(256)
void convT_bf16(const float* __restrict__ W, __bf16* __restrict__ Wt,
                int rows, int cols) {
    int i = blockIdx.x * 256 + threadIdx.x;          // i < rows*cols, indexes Wt
    if (i >= rows * cols) return;
    int n = i / rows, k = i - n * rows;
    Wt[i] = (__bf16)W[k * cols + n];
}

// ---------------- WMMA GEMM: C[M x NCOLS] = A[M x K](bf16,row) * Bt[NCOLS x K](bf16) ----------------
template<int K, int NCOLS>
__global__ __launch_bounds__(256)
void gemm_bf16_wmma(const __bf16* __restrict__ A, const __bf16* __restrict__ Bt,
                    float* __restrict__ C, int M) {
    constexpr int TN = NCOLS / 16;
    int wave = blockIdx.x * (blockDim.x >> 5) + (threadIdx.x >> 5);
    int lane = threadIdx.x & 31;
    int tm = wave / TN, tn = wave - tm * TN;
    if (tm * 16 >= M) return;                         // wave-uniform: EXEC stays all-1s
    int l16 = lane & 15, lh = lane >> 4;

    const __bf16* arow = A  + (size_t)(tm * 16 + l16) * K;   // lane row M = l16
    const __bf16* brow = Bt + (size_t)(tn * 16 + l16) * K;   // lane col N = l16

    v8f acc = {};
#pragma unroll
    for (int kb = 0; kb < K; kb += 32) {
        v16bf a, b;
        // A lane layout: lh==0 -> K{kb..kb+7, kb+16..kb+23}; lh==1 -> +8 shifted
        ((uint4*)&a)[0] = *(const uint4*)(arow + kb + lh * 8);
        ((uint4*)&a)[1] = *(const uint4*)(arow + kb + lh * 8 + 16);
        // B lane layout: same as A: lh==0 -> K{kb..kb+7, kb+16..kb+23}; lh==1 -> +8 shifted
        ((uint4*)&b)[0] = *(const uint4*)(brow + kb + lh * 8);
        ((uint4*)&b)[1] = *(const uint4*)(brow + kb + lh * 8 + 16);
        acc = __builtin_amdgcn_wmma_f32_16x16x32_bf16(false, a, false, b,
                                                      (short)0, acc, false, false);
    }
    // C/D layout: VGPR r -> row (tm*16 + lh*8 + r), col (tn*16 + l16)
    float* crow = C + (size_t)(tm * 16 + lh * 8) * NCOLS + tn * 16 + l16;
#pragma unroll
    for (int r = 0; r < 8; ++r) crow[(size_t)r * NCOLS] = acc[r];
}

// ---------------- per-node attention logits ----------------
// asrc[n*H+h] = dot(h[n,h,:], a_src[h,:]) ; same for adst
__global__ __launch_bounds__(256)
void alpha_nodes(const float* __restrict__ h, const float* __restrict__ a_src,
                 const float* __restrict__ a_dst, float* __restrict__ asrc,
                 float* __restrict__ adst, int H, int C) {
    int t = blockIdx.x * 256 + threadIdx.x;          // t < N*H
    if (t >= N_NODES * H) return;
    int n = t / H, hh = t - n * H;
    const float* hp = h + (size_t)n * H * C + hh * C;
    const float* sp = a_src + hh * C;
    const float* dp = a_dst + hh * C;
    float s = 0.f, d = 0.f;
    for (int c = 0; c < C; ++c) { float v = hp[c]; s += v * sp[c]; d += v * dp[c]; }
    asrc[t] = s; adst[t] = d;
}

// ---------------- edge passes ----------------
// pass 1: e = leaky_relu(asrc[src]+adst[dst]); store; segment max over dst
__global__ __launch_bounds__(256)
void edge_max(const int* __restrict__ ei, const float* __restrict__ asrc,
              const float* __restrict__ adst, float* __restrict__ ew,
              float* __restrict__ m, int H) {
    int t = blockIdx.x * 256 + threadIdx.x;          // t < EL*H
    if (t >= EL * H) return;
    int e = t / H, hh = t - e * H;
    int s, d; edge_sd(ei, e, s, d);
    float v = asrc[s * H + hh] + adst[d * H + hh];
    v = (v > 0.f) ? v : 0.2f * v;
    ew[t] = v;
    atomicMaxF(&m[d * H + hh], v);
}

// pass 2: ex = exp(e - m[dst]); store; segment sum over dst
__global__ __launch_bounds__(256)
void edge_expsum(const int* __restrict__ ei, float* __restrict__ ew,
                 const float* __restrict__ m, float* __restrict__ den, int H) {
    int t = blockIdx.x * 256 + threadIdx.x;
    if (t >= EL * H) return;
    int e = t / H, hh = t - e * H;
    int s, d; edge_sd(ei, e, s, d);
    float ex = __expf(ew[t] - m[d * H + hh]);
    ew[t] = ex;
    atomicAdd(&den[d * H + hh], ex);
}

// pass 3: out[dst] += h[src] * (ex / (den[dst]+1e-16)); EPB edges per 256-thread block
__global__ __launch_bounds__(256)
void edge_aggr(const int* __restrict__ ei, const float* __restrict__ h,
               const float* __restrict__ ew, const float* __restrict__ den,
               float* __restrict__ out, int H, int C, int EPB) {
    int hc = H * C;
    int el = threadIdx.x / hc;
    int e  = blockIdx.x * EPB + el;
    if (e >= EL) return;
    int j  = threadIdx.x - el * hc;
    int hh = j / C;
    int s, d; edge_sd(ei, e, s, d);
    float alpha = ew[e * H + hh] / (den[d * H + hh] + 1e-16f);
    atomicAdd(&out[(size_t)d * hc + j], h[(size_t)s * hc + j] * alpha);
}

// ---------------- epilogues ----------------
// x2 = bf16( elu(out1 + b1) )
__global__ __launch_bounds__(256)
void finalize1(const float* __restrict__ out1, const float* __restrict__ b1,
               __bf16* __restrict__ x2) {
    int i = blockIdx.x * 256 + threadIdx.x;          // i < N*HC1
    float v = out1[i] + b1[i & (HC1 - 1)];
    v = (v > 0.f) ? v : (__expf(v) - 1.0f);
    x2[i] = (__bf16)v;
}

// out = out2 + b2
__global__ __launch_bounds__(256)
void finalize2(const float* __restrict__ out2, const float* __restrict__ b2,
               float* __restrict__ out) {
    int i = blockIdx.x * 256 + threadIdx.x;          // i < N*HIDC
    out[i] = out2[i] + b2[i & (HIDC - 1)];
}

// ---------------- launch ----------------
extern "C" void kernel_launch(void* const* d_in, const int* in_sizes, int n_in,
                              void* d_out, int out_size, void* d_ws, size_t ws_size,
                              hipStream_t stream) {
    const float* x   = (const float*)d_in[0];
    const int*   ei  = (const int*)  d_in[1];
    const float* pe  = (const float*)d_in[2];
    const float* W1  = (const float*)d_in[3];
    const float* as1 = (const float*)d_in[4];
    const float* ad1 = (const float*)d_in[5];
    const float* b1  = (const float*)d_in[6];
    const float* W2  = (const float*)d_in[7];
    const float* as2 = (const float*)d_in[8];
    const float* ad2 = (const float*)d_in[9];
    const float* b2  = (const float*)d_in[10];
    float* out = (float*)d_out;

    // ---- carve workspace (each sub-buffer 256B aligned) ----
    unsigned char* w = (unsigned char*)d_ws;
    size_t off = 0;
    auto carve = [&](size_t bytes) -> void* {
        void* p = w + off;
        off += (bytes + 255) & ~((size_t)255);
        return p;
    };
    __bf16* h0bf  = (__bf16*)carve((size_t)N_NODES * DIM * 2);
    __bf16* W1t   = (__bf16*)carve((size_t)HC1 * DIM * 2);       // [256][128]
    __bf16* W2t   = (__bf16*)carve((size_t)HIDC * HC1 * 2);      // [64][256]
    float*  h1    = (float*) carve((size_t)N_NODES * HC1 * 4);
    float*  asrc1 = (float*) carve((size_t)N_NODES * 4 * 4);
    float*  adst1 = (float*) carve((size_t)N_NODES * 4 * 4);
    float*  m1    = (float*) carve((size_t)N_NODES * 4 * 4);
    float*  den1  = (float*) carve((size_t)N_NODES * 4 * 4);
    float*  ew1   = (float*) carve((size_t)EL * 4 * 4);
    float*  out1  = (float*) carve((size_t)N_NODES * HC1 * 4);
    __bf16* x2bf  = (__bf16*)carve((size_t)N_NODES * HC1 * 2);
    float*  h2    = (float*) carve((size_t)N_NODES * HIDC * 4);
    float*  asrc2 = (float*) carve((size_t)N_NODES * 4);
    float*  adst2 = (float*) carve((size_t)N_NODES * 4);
    float*  m2    = (float*) carve((size_t)N_NODES * 4);
    float*  den2  = (float*) carve((size_t)N_NODES * 4);
    float*  ew2   = (float*) carve((size_t)EL * 4);
    float*  out2  = (float*) carve((size_t)N_NODES * HIDC * 4);

    // ---- prep: PE add + bf16 conversions ----
    prep_h0  <<<(N_NODES * DIM) / 256, 256, 0, stream>>>(x, pe, h0bf);
    convT_bf16<<<(DIM * HC1 + 255) / 256, 256, 0, stream>>>(W1, W1t, DIM, HC1);
    convT_bf16<<<(HC1 * HIDC + 255) / 256, 256, 0, stream>>>(W2, W2t, HC1, HIDC);

    // ---- layer 1 ----
    // GEMM1: h1 = h0 @ W1   (16384x128x256), one 16x16 tile per wave, 8 waves/block
    {
        int tiles = (N_NODES / 16) * (HC1 / 16);
        gemm_bf16_wmma<DIM, HC1><<<tiles / 8, 256, 0, stream>>>(h0bf, W1t, h1, N_NODES);
    }
    alpha_nodes<<<(N_NODES * 4) / 256, 256, 0, stream>>>(h1, as1, ad1, asrc1, adst1, 4, HIDC);

    fill_f32<<<(N_NODES * 4) / 256, 256, 0, stream>>>(m1,  -INFINITY, N_NODES * 4);
    fill_f32<<<(N_NODES * 4) / 256, 256, 0, stream>>>(den1, 0.0f,     N_NODES * 4);
    fill_f32<<<(N_NODES * HC1) / 256, 256, 0, stream>>>(out1, 0.0f,   N_NODES * HC1);

    edge_max   <<<(EL * 4 + 255) / 256, 256, 0, stream>>>(ei, asrc1, adst1, ew1, m1, 4);
    edge_expsum<<<(EL * 4 + 255) / 256, 256, 0, stream>>>(ei, ew1, m1, den1, 4);
    edge_aggr  <<<EL, 256, 0, stream>>>(ei, h1, ew1, den1, out1, 4, HIDC, 1);

    finalize1<<<(N_NODES * HC1) / 256, 256, 0, stream>>>(out1, b1, x2bf);

    // ---- layer 2 ----
    // GEMM2: h2 = x2 @ W2   (16384x256x64)
    {
        int tiles = (N_NODES / 16) * (HIDC / 16);
        gemm_bf16_wmma<HC1, HIDC><<<tiles / 8, 256, 0, stream>>>(x2bf, W2t, h2, N_NODES);
    }
    alpha_nodes<<<N_NODES / 256, 256, 0, stream>>>(h2, as2, ad2, asrc2, adst2, 1, HIDC);

    fill_f32<<<N_NODES / 256, 256, 0, stream>>>(m2,  -INFINITY, N_NODES);
    fill_f32<<<N_NODES / 256, 256, 0, stream>>>(den2, 0.0f,     N_NODES);
    fill_f32<<<(N_NODES * HIDC) / 256, 256, 0, stream>>>(out2, 0.0f, N_NODES * HIDC);

    edge_max   <<<(EL + 255) / 256, 256, 0, stream>>>(ei, asrc2, adst2, ew2, m2, 1);
    edge_expsum<<<(EL + 255) / 256, 256, 0, stream>>>(ei, ew2, m2, den2, 1);
    edge_aggr  <<<EL / 4, 256, 0, stream>>>(ei, h2, ew2, den2, out2, 1, HIDC, 4);

    finalize2<<<(N_NODES * HIDC) / 256, 256, 0, stream>>>(out2, b2, out);
}
